// ALBEF_12292196401889
// MI455X (gfx1250) — matrix operand order repaired and checked by reading, not compiled, not run on hardware
//
#include <hip/hip_runtime.h>
#include <stdint.h>

typedef __bf16 bf16_t;
typedef bf16_t v16bf __attribute__((ext_vector_type(16)));
typedef float  v8f   __attribute__((ext_vector_type(8)));

union Frag16 {
    unsigned short s[16];
    uint4          q[2];
    v16bf          v;
};

__device__ __forceinline__ unsigned short f2bf(float f) {
    unsigned int u = __float_as_uint(f);
    unsigned int r = (u + 0x7FFFu + ((u >> 16) & 1u)) >> 16;   // RNE
    return (unsigned short)r;
}

static constexpr int BB = 256;   // batch
static constexpr int Q  = 32;    // concept queries
static constexpr int LL = 40;    // text seq len
static constexpr int LP = 48;    // padded text seq len (3 n-tiles of 16)
static constexpr int VW = 768;   // input width (both)
static constexpr int DD = 256;   // embed dim

// ---------------------------------------------------------------- weights ---
// WT[n][k] = bf16(W[k][n]) : makes WMMA B-fragments contiguous per lane.
__global__ void k_wt(const float* __restrict__ W, unsigned short* __restrict__ WT) {
    int idx = blockIdx.x * blockDim.x + threadIdx.x;
    if (idx >= VW * DD) return;
    int k = idx >> 8, n = idx & 255;          // coalesced read of W[k][n]
    WT[(size_t)n * VW + k] = f2bf(W[idx]);
}

// -------------------------------------------------------- X f32 -> bf16 -----
// One streaming conversion pass so k_proj A-fragments are plain b128 loads
// (removes the redundant per-wave RNE VALU work seen in round 1 asm).
__global__ void k_xb(const float* __restrict__ X, unsigned short* __restrict__ O, int n8) {
    int idx = blockIdx.x * blockDim.x + threadIdx.x;
    if (idx >= n8) return;
    const float4* p = (const float4*)(X + (size_t)idx * 8);
    float4 f0 = p[0], f1 = p[1];
    union { unsigned short s[8]; uint4 q; } o;
    o.s[0] = f2bf(f0.x); o.s[1] = f2bf(f0.y); o.s[2] = f2bf(f0.z); o.s[3] = f2bf(f0.w);
    o.s[4] = f2bf(f1.x); o.s[5] = f2bf(f1.y); o.s[6] = f2bf(f1.z); o.s[7] = f2bf(f1.w);
    ((uint4*)O)[idx] = o.q;
}

// ----------------------------------------------------------------- lengths --
// masks are contiguous [1,1,...,1,0,...]; after CLS/SEP removal valid t in [1, len-2]
__global__ void k_len(const int* __restrict__ mask, int* __restrict__ nw) {
    int j = threadIdx.x;
    if (j < BB) {
        int s = 0;
        for (int t = 0; t < LL; ++t) s += (mask[j * LL + t] != 0) ? 1 : 0;
        nw[j] = s - 2;
    }
}

// -------------------------------------------------- projection + L2 norm ----
// Block: 256 threads = 8 waves. Block computes 16 padded rows x 256 cols.
// Wave w covers cols [32w, 32w+32). K = 768 in 24 steps of 32 (bf16 WMMA),
// double-buffered A/B fragments.
__global__ __launch_bounds__(256)
void k_proj(const unsigned short* __restrict__ Xb, const unsigned short* __restrict__ WT,
            const float* __restrict__ bias, unsigned short* __restrict__ out,
            int in_seq, int out_seq) {
    __shared__ float Y[16 * 256];
    __shared__ float part[16 * 16];
    __shared__ float rn[16];

    const int tid = threadIdx.x;
    const int w   = tid >> 5;
    const int l   = tid & 31;
    const int ln  = l & 15;
    const int hi  = (l < 16) ? 0 : 1;
    const int pr0 = blockIdx.x * 16;

    // A-fragment row for this lane
    const int  prA = pr0 + ln;
    const int  bA  = prA / out_seq;
    const int  tA  = prA % out_seq;
    const bool rowValid = (tA < in_seq);
    const unsigned short* xr = Xb + (size_t)(bA * in_seq + tA) * VW;

    const int aoff  = hi ? 8 : 0;     // A: K sub-offset per lane half
    const int koffB = hi ? 8 : 0;     // B: K sub-offset per lane half
    const int ncol0 = w * 32 + ln;
    const unsigned short* wtb0 = WT + (size_t)ncol0 * VW + koffB;
    const unsigned short* wtb1 = WT + (size_t)(ncol0 + 16) * VW + koffB;

    v8f acc[2];
    for (int nt = 0; nt < 2; ++nt)
        for (int e = 0; e < 8; ++e) acc[nt][e] = 0.0f;

    Frag16 aq[2], bq[2][2];
    // preload k-step 0
    if (rowValid) {
        aq[0].q[0] = *(const uint4*)(xr + aoff);
        aq[0].q[1] = *(const uint4*)(xr + aoff + 16);
    } else {
        aq[0].q[0] = make_uint4(0, 0, 0, 0);
        aq[0].q[1] = make_uint4(0, 0, 0, 0);
    }
    bq[0][0].q[0] = *(const uint4*)(wtb0);
    bq[0][0].q[1] = *(const uint4*)(wtb0 + 16);
    bq[0][1].q[0] = *(const uint4*)(wtb1);
    bq[0][1].q[1] = *(const uint4*)(wtb1 + 16);

    for (int ks = 0; ks < VW / 32; ++ks) {
        const int cur = ks & 1, nxt = cur ^ 1;
        if (ks < VW / 32 - 1) {
            const int kn = (ks + 1) * 32;
            if (rowValid) {
                aq[nxt].q[0] = *(const uint4*)(xr + kn + aoff);
                aq[nxt].q[1] = *(const uint4*)(xr + kn + aoff + 16);
            } else {
                aq[nxt].q[0] = make_uint4(0, 0, 0, 0);
                aq[nxt].q[1] = make_uint4(0, 0, 0, 0);
            }
            bq[nxt][0].q[0] = *(const uint4*)(wtb0 + kn);
            bq[nxt][0].q[1] = *(const uint4*)(wtb0 + kn + 16);
            bq[nxt][1].q[0] = *(const uint4*)(wtb1 + kn);
            bq[nxt][1].q[1] = *(const uint4*)(wtb1 + kn + 16);
        }
        acc[0] = __builtin_amdgcn_wmma_f32_16x16x32_bf16(
            false, aq[cur].v, false, bq[cur][0].v, (short)0, acc[0], false, false);
        acc[1] = __builtin_amdgcn_wmma_f32_16x16x32_bf16(
            false, aq[cur].v, false, bq[cur][1].v, (short)0, acc[1], false, false);
    }

    // bias + dump to LDS (C layout: VGPR e <-> row e + 8*hi, lane%16 <-> col)
    for (int nt = 0; nt < 2; ++nt) {
        const int n  = ncol0 + nt * 16;
        const float bv = bias[n];
        for (int e = 0; e < 8; ++e)
            Y[(e + hi * 8) * 256 + n] = acc[nt][e] + bv;
    }
    __syncthreads();

    {   // per-row sum of squares
        int row = tid >> 4, seg = tid & 15;
        float s = 0.0f;
        for (int c = 0; c < 16; ++c) {
            float v = Y[row * 256 + seg * 16 + c];
            s += v * v;
        }
        part[row * 16 + seg] = s;
    }
    __syncthreads();
    if (tid < 16) {
        float s = 0.0f;
        for (int c = 0; c < 16; ++c) s += part[tid * 16 + c];
        rn[tid] = (s > 0.0f) ? rsqrtf(s) : 0.0f;
    }
    __syncthreads();

    {   // normalized bf16 writeback (padded rows -> zeros)
        int row = tid >> 4, seg = tid & 15;
        int pr  = pr0 + row;
        int t   = pr % out_seq;
        float r = (t < in_seq) ? rn[row] : 0.0f;
        Frag16 o;
        for (int c = 0; c < 16; ++c)
            o.s[c] = f2bf(Y[row * 256 + seg * 16 + c] * r);
        uint4* op = (uint4*)(out + (size_t)pr * 256 + seg * 16);
        op[0] = o.q[0];
        op[1] = o.q[1];
    }
}

// ------------------------------------------------------ pairwise similarity -
// Block i: stage cf[i] (32x256 bf16, 16KB) into LDS with async global->LDS
// copies (ASYNCcnt path); 4 waves sweep j with double-buffered B fragments.
// Per (i,j): 2x3 C tiles over K=256 -> 48 WMMAs; reduce masked row/col maxes.
__global__ __launch_bounds__(128)
void k_sim(const unsigned short* __restrict__ cfn,
           const unsigned short* __restrict__ wfn,
           const int* __restrict__ nwords, float* __restrict__ sim) {
    __shared__ unsigned short cfl[Q * DD];   // 16 KB
    __shared__ float Sb[4][Q * LP];          // 24 KB (per-wave 32x48 tile)

    const int i   = blockIdx.x;
    const int tid = threadIdx.x;
    const int w   = tid >> 5;
    const int l   = tid & 31;
    const int ln  = l & 15;
    const int hi  = (l < 16) ? 0 : 1;
    const int aoff  = hi ? 8 : 0;
    const int koffB = hi ? 8 : 0;

    {   // async stage of cf[i]: flat LDS addr low 32 bits == LDS byte offset
        const unsigned lbase = (unsigned)(size_t)(void*)cfl;
        const unsigned short* src = cfn + (size_t)i * Q * DD;
        for (int idx = tid; idx < (Q * DD) / 8; idx += 128) {
            unsigned long long ga = (unsigned long long)(size_t)(src + idx * 8);
            unsigned la = lbase + (unsigned)idx * 16u;
            asm volatile("global_load_async_to_lds_b128 %0, %1, off"
                         :: "v"(la), "v"(ga) : "memory");
        }
        asm volatile("s_wait_asynccnt 0x0" ::: "memory");
    }
    __syncthreads();

    float* Sw = &Sb[w][0];

    for (int jt = 0; jt < BB / 4; ++jt) {
        const int j  = jt * 4 + w;
        const int nw = nwords[j];
        const unsigned short* wj = wfn + (size_t)j * LP * DD + koffB;

        v8f acc[2][3];
        for (int mt = 0; mt < 2; ++mt)
            for (int nt = 0; nt < 3; ++nt)
                for (int e = 0; e < 8; ++e) acc[mt][nt][e] = 0.0f;

        Frag16 bq[2][3];
        for (int nt = 0; nt < 3; ++nt) {          // preload k-step 0
            const unsigned short* p = wj + (size_t)(nt * 16 + ln) * DD;
            bq[0][nt].q[0] = *(const uint4*)(p);
            bq[0][nt].q[1] = *(const uint4*)(p + 16);
        }

        for (int ks = 0; ks < DD / 32; ++ks) {
            const int cur = ks & 1, nxt = cur ^ 1;
            const int k0 = ks * 32;
            if (ks < DD / 32 - 1) {
                const int kn = k0 + 32;
                for (int nt = 0; nt < 3; ++nt) {
                    const unsigned short* p = wj + (size_t)(nt * 16 + ln) * DD + kn;
                    bq[nxt][nt].q[0] = *(const uint4*)(p);
                    bq[nxt][nt].q[1] = *(const uint4*)(p + 16);
                }
            }
            Frag16 a[2];
            for (int mt = 0; mt < 2; ++mt) {
                const int m = mt * 16 + ln;
                a[mt].q[0] = *(const uint4*)(cfl + m * DD + k0 + aoff);
                a[mt].q[1] = *(const uint4*)(cfl + m * DD + k0 + aoff + 16);
            }
            for (int mt = 0; mt < 2; ++mt)
                for (int nt = 0; nt < 3; ++nt)
                    acc[mt][nt] = __builtin_amdgcn_wmma_f32_16x16x32_bf16(
                        false, a[mt].v, false, bq[cur][nt].v, (short)0, acc[mt][nt], false, false);
        }

        // dump 32x48 score tile to this wave's LDS region
        for (int mt = 0; mt < 2; ++mt)
            for (int nt = 0; nt < 3; ++nt)
                for (int e = 0; e < 8; ++e) {
                    const int m = mt * 16 + e + hi * 8;
                    const int n = nt * 16 + ln;
                    Sw[m * LP + n] = acc[mt][nt][e];
                }
        __syncthreads();

        // row term: for each valid t, max over all 32 s (c_mask all ones)
        float sum_row = 0.0f;
        {
            const int t = l;
            float rm = -1e30f;
            for (int m = 0; m < Q; ++m) rm = fmaxf(rm, Sw[m * LP + t]);
            if (t >= 1 && t <= nw) sum_row += rm;
            if (l < 16) {
                const int t2 = 32 + l;
                float rm2 = -1e30f;
                for (int m = 0; m < Q; ++m) rm2 = fmaxf(rm2, Sw[m * LP + t2]);
                if (t2 <= nw) sum_row += rm2;
            }
        }
        // col term: for each s (=lane), max over valid t in [1, nw]
        float cm = -1e30f;
        {
            const int m = l;
            for (int t = 1; t <= nw; ++t) cm = fmaxf(cm, Sw[m * LP + t]);
        }
        float sr = sum_row, sc = cm;
        for (int off = 16; off >= 1; off >>= 1) {
            sr += __shfl_xor(sr, off, 32);
            sc += __shfl_xor(sc, off, 32);
        }
        if (l == 0)
            sim[i * BB + j] = sr / (float)nw + sc * (1.0f / (float)Q);
        __syncthreads();
    }
}

// ---------------------------------------------------------------- loss ------
// sim_t2i == sim_i2t^T, so both losses come from row/col logsumexp of one matrix.
__global__ __launch_bounds__(256)
void k_loss(const float* __restrict__ sim, const float* __restrict__ temp,
            float* __restrict__ out) {
    __shared__ float red[256];
    const int i = threadIdx.x;
    const float invT = 1.0f / temp[0];

    const float diag = sim[i * BB + i] * invT;
    float rmax = -1e30f, cmax = -1e30f;
    for (int j = 0; j < BB; ++j) {
        rmax = fmaxf(rmax, sim[i * BB + j] * invT);
        cmax = fmaxf(cmax, sim[j * BB + i] * invT);
    }
    float rs = 0.0f, cs = 0.0f;
    for (int j = 0; j < BB; ++j) {
        rs += __expf(sim[i * BB + j] * invT - rmax);
        cs += __expf(sim[j * BB + i] * invT - cmax);
    }
    const float rlse = rmax + __logf(rs);
    const float clse = cmax + __logf(cs);
    red[i] = (diag - rlse) + (diag - clse);
    __syncthreads();
    for (int s = 128; s >= 1; s >>= 1) {
        if (i < s) red[i] += red[i + s];
        __syncthreads();
    }
    if (i == 0) out[0] = -red[0] / (2.0f * (float)BB);
}

// ---------------------------------------------------------------- driver ----
extern "C" void kernel_launch(void* const* d_in, const int* in_sizes, int n_in,
                              void* d_out, int out_size, void* d_ws, size_t ws_size,
                              hipStream_t stream) {
    (void)in_sizes; (void)n_in; (void)out_size; (void)ws_size;

    const float* concept_feat = (const float*)d_in[0];
    const float* text_embeds  = (const float*)d_in[1];
    const int*   text_mask    = (const int*)d_in[2];
    const float* Wc           = (const float*)d_in[3];
    const float* bc           = (const float*)d_in[4];
    const float* Ww           = (const float*)d_in[5];
    const float* bw           = (const float*)d_in[6];
    const float* temp_cpt     = (const float*)d_in[7];
    float*       out          = (float*)d_out;

    char*  ws  = (char*)d_ws;
    size_t off = 0;
    auto carve = [&](size_t bytes) -> void* {
        off = (off + 255) & ~(size_t)255;
        void* p = ws + off;
        off += bytes;
        return p;
    };
    unsigned short* WcT   = (unsigned short*)carve((size_t)VW * DD * 2);
    unsigned short* WwT   = (unsigned short*)carve((size_t)VW * DD * 2);
    unsigned short* Xcb   = (unsigned short*)carve((size_t)BB * Q  * VW * 2);
    unsigned short* Xwb   = (unsigned short*)carve((size_t)BB * LL * VW * 2);
    unsigned short* cfn   = (unsigned short*)carve((size_t)BB * Q  * DD * 2);
    unsigned short* wfn   = (unsigned short*)carve((size_t)BB * LP * DD * 2);
    float*          sim   = (float*)carve((size_t)BB * BB * 4);
    int*            nwrds = (int*)carve((size_t)BB * 4);

    const int n8c = (BB * Q  * VW) / 8;
    const int n8w = (BB * LL * VW) / 8;

    k_wt  <<<(VW * DD) / 256, 256, 0, stream>>>(Wc, WcT);
    k_wt  <<<(VW * DD) / 256, 256, 0, stream>>>(Ww, WwT);
    k_xb  <<<(n8c + 255) / 256, 256, 0, stream>>>(concept_feat, Xcb, n8c);
    k_xb  <<<(n8w + 255) / 256, 256, 0, stream>>>(text_embeds,  Xwb, n8w);
    k_len <<<1, 256, 0, stream>>>(text_mask, nwrds);
    k_proj<<<(BB * Q)  / 16, 256, 0, stream>>>(Xcb, WcT, bc, cfn, Q,  Q);
    k_proj<<<(BB * LP) / 16, 256, 0, stream>>>(Xwb, WwT, bw, wfn, LL, LP);
    k_sim <<<BB, 128, 0, stream>>>(cfn, wfn, nwrds, sim);
    k_loss<<<1, 256, 0, stream>>>(sim, temp_cpt, out);
}
